// HurstEstimator_80874234184106
// MI455X (gfx1250) — matrix operand order repaired and checked by reading, not compiled, not run on hardware
//
#include <hip/hip_runtime.h>

typedef __attribute__((ext_vector_type(16))) _Float16 v16h;
typedef __attribute__((ext_vector_type(8)))  float    v8f;
typedef __attribute__((ext_vector_type(2)))  __fp16   h2v;

#define BATCH 1024
#define TLEN  4096
#define WSZ   10
#define HIDN  32
#define NWIN  (TLEN - WSZ)        // 4086 valid windows per row
#define CHUNK 512                 // windows per block
#define NCHUNK (TLEN / CHUNK)     // 8 chunks cover 4096 >= 4086 (tail masked)
#define WAVES 8
#define TILES 4                   // 16-window tiles per wave (processed in pairs)

// single v_med3_f32, immune to clang's fmax canonicalization
__device__ __forceinline__ float relu_(float x) {
    float y;
    asm("v_med3_f32 %0, %1, 0, 0x7f800000" : "=v"(y) : "v"(x));
    return y;
}
__device__ __forceinline__ unsigned pkh2(float a, float b) {
    h2v h = __builtin_amdgcn_cvt_pkrtz(a, b);                   // v_cvt_pk_rtz_f16_f32
    union { h2v h; unsigned u; } c; c.h = h; return c.u;
}
// sigma permutation for layer-2 K reorder (see LDS interleaved layout below)
__device__ __forceinline__ int sigk(int K) {
    const int a = K >> 3, b = K & 7;
    return (b >> 1) + ((b & 1) << 4) + ((a & 1) << 3) + ((a >> 1) << 2);
}

__launch_bounds__(256)
__global__ void hurst_mlp_wmma(const float* __restrict__ ret,
                               const float* __restrict__ W1,
                               const float* __restrict__ b1,
                               const float* __restrict__ W2,
                               const float* __restrict__ b2,
                               const float* __restrict__ W3,
                               const float* __restrict__ b3,
                               float* __restrict__ out)
{
    __shared__ __align__(16) float    sRetF[CHUNK + 16];        // chunk + halo (f32)
    // weights pre-staged in per-lane B-operand order: [((g*2+hbit)*16+n)*16 + j]
    __shared__ __align__(16) _Float16 sW1p[1024];               // layer1 (b1 folded at K=10)
    __shared__ __align__(16) _Float16 sW2p[1024];               // layer2 (sigma-permuted rows)
    __shared__ float sB2[HIDN], sW3[HIDN];
    __shared__ __align__(16) _Float16 sH1[WAVES * 2 * 16 * 40]; // h1 bounce, dbl-buffered
    __shared__ __align__(16) float    sP[WAVES * 2 * 16 * 18];  // layer3 transpose-reduce

    const int tid   = threadIdx.x;
    const int row   = blockIdx.x >> 3;
    const int chunk = blockIdx.x & (NCHUNK - 1);
    const int lane  = tid & 31;
    const int wave  = tid >> 5;
    const int n0    = lane & 15;
    const bool hi   = lane >= 16;

    // ---- stage returns chunk (index-clamped; tail windows are store-masked) ----
    const float* r0 = ret + (size_t)row * TLEN;
    for (int i = tid; i < CHUNK + 16; i += 256) {
        int g = chunk * CHUNK + i;
        g = (g < TLEN) ? g : (TLEN - 1);
        sRetF[i] = r0[g];
    }
    // ---- stage weights directly in per-lane B layout (scatter once, read wide) ----
    for (int idx = tid; idx < 1024; idx += 256) {
        const int j  = idx & 15;
        const int n  = (idx >> 4) & 15;
        const int hb = (idx >> 8) & 1;
        const int g  = idx >> 9;
        const int K  = (j < 8) ? (hb * 8 + j) : (16 + hb * 8 + (j - 8));
        const int col = 16 * g + n;
        _Float16 w1v;
        if (K < WSZ)       w1v = (_Float16)W1[K * HIDN + col];
        else if (K == WSZ) w1v = (_Float16)b1[col];     // bias row (A supplies 1.0)
        else               w1v = (_Float16)0;
        sW1p[idx] = w1v;
        sW2p[idx] = (_Float16)W2[sigk(K) * HIDN + col];
    }
    if (tid < HIDN) { sB2[tid] = b2[tid]; sW3[tid] = W3[tid]; }
    __syncthreads();

    const float b3v = b3[0];

    // ---- B operands: 8x ds_load_b128, held in VGPRs for the whole kernel ----
    v16h B1t[2], B2t[2];
#pragma unroll
    for (int g = 0; g < 2; ++g) {
        const int base = ((g * 2 + (hi ? 1 : 0)) * 16 + n0) * 16;
        union { v16h v; uint4 q[2]; } t;
        const uint4* q1 = reinterpret_cast<const uint4*>(&sW1p[base]);
        t.q[0] = q1[0]; t.q[1] = q1[1];
        B1t[g] = t.v;
        const uint4* q2 = reinterpret_cast<const uint4*>(&sW2p[base]);
        t.q[0] = q2[0]; t.q[1] = q2[1];
        B2t[g] = t.v;
    }
    const float bias20 = sB2[n0], bias21 = sB2[n0 + 16];
    const float w3a = sW3[n0],    w3b = sW3[n0 + 16];

    _Float16* const wH1 = &sH1[wave * 2 * 16 * 40];
    float*    const wP  = &sP[wave * 2 * 16 * 18];
    const int Mofs = hi ? 8 : 0;
    const unsigned msk = hi ? 0u : ~0u;

#pragma unroll 1
    for (int tp = 0; tp < TILES / 2; ++tp) {
        // two tiles (p = 0,1) software-pipelined; independent LDS buffers per p
        union { v16h v; unsigned u[8]; } au[2];
        v8f c0[2], c1[2];

        // ---- stage 1: build both A1 operands (branch-free, packed cvt) ----
        // lo lanes: K=0..7 ; hi lanes: K=8,9 plus constant 1.0 at K=10 (bias row)
#pragma unroll
        for (int p = 0; p < 2; ++p) {
            const int tw = wave * (TILES * 16) + (2 * tp + p) * 16;
            const float* fp = sRetF + (tw + n0) + (hi ? 8 : 0);
            const float f0 = fp[0], f1 = fp[1], f2 = fp[2], f3 = fp[3];
            const float f4 = fp[4], f5 = fp[5], f6 = fp[6], f7 = fp[7];
            au[p].u[0] = pkh2(f0, f1);
            au[p].u[1] = (pkh2(f2, f3) & msk) | (~msk & 0x00003C00u);
            au[p].u[2] = pkh2(f4, f5) & msk;
            au[p].u[3] = pkh2(f6, f7) & msk;
            au[p].u[4] = 0u; au[p].u[5] = 0u; au[p].u[6] = 0u; au[p].u[7] = 0u;
        }
        // ---- stage 2: four layer-1 WMMAs back-to-back, C = inline 0 ----
#pragma unroll
        for (int p = 0; p < 2; ++p) {
            const v8f zz = {};
            c0[p] = __builtin_amdgcn_wmma_f32_16x16x32_f16(false, au[p].v, false, B1t[0], (short)0, zz, false, false);
            c1[p] = __builtin_amdgcn_wmma_f32_16x16x32_f16(false, au[p].v, false, B1t[1], (short)0, zz, false, false);
        }
        // ---- stage 3: ReLU + pack + LDS bounce (interleaved c=2n+g, stride 40 halfs) ----
#pragma unroll
        for (int p = 0; p < 2; ++p) {
            _Float16* myH1 = wH1 + p * (16 * 40);
#pragma unroll
            for (int r = 0; r < 8; ++r) {
                const unsigned pk = pkh2(relu_(c0[p][r]), relu_(c1[p][r]));
                *reinterpret_cast<unsigned*>(myH1 + (Mofs + r) * 40 + 2 * n0) = pk;
            }
        }
        union { v16h v; uint4 q[2]; } a2[2];
        v8f d0[2], d1[2];
#pragma unroll
        for (int p = 0; p < 2; ++p) {
            const uint4* p4 = reinterpret_cast<const uint4*>(wH1 + p * (16 * 40) + n0 * 40 + (hi ? 16 : 0));
            a2[p].q[0] = p4[0];
            a2[p].q[1] = p4[1];
#pragma unroll
            for (int r = 0; r < 8; ++r) { d0[p][r] = bias20; d1[p][r] = bias21; }
        }
        // ---- stage 4: four layer-2 WMMAs back-to-back ----
#pragma unroll
        for (int p = 0; p < 2; ++p) {
            d0[p] = __builtin_amdgcn_wmma_f32_16x16x32_f16(false, a2[p].v, false, B2t[0], (short)0, d0[p], false, false);
            d1[p] = __builtin_amdgcn_wmma_f32_16x16x32_f16(false, a2[p].v, false, B2t[1], (short)0, d1[p], false, false);
        }
        // ---- stage 5: layer 3 via transpose-reduce through LDS ----
        // partial(M,n) = relu(h2[M,n])*W3[n] + relu(h2[M,n+16])*W3[n+16]
#pragma unroll
        for (int p = 0; p < 2; ++p) {
            float* myP = wP + p * (16 * 18);
#pragma unroll
            for (int r = 0; r < 8; ++r)
                myP[(Mofs + r) * 18 + n0] = relu_(d0[p][r]) * w3a + relu_(d1[p][r]) * w3b;
            // lane L sums 8 partials of row M = L&15 (half chosen by L/16), then
            // combines wave halves with one xor-16 shuffle
            const float* rp = myP + n0 * 18 + (hi ? 8 : 0);
            float s = ((rp[0] + rp[1]) + (rp[2] + rp[3])) + ((rp[4] + rp[5]) + (rp[6] + rp[7]));
            s += __shfl_xor(s, 16);
            const float val = 0.5f / (1.f + __expf(-(s + b3v)));
            const int tw = wave * (TILES * 16) + (2 * tp + p) * 16;
            const int w  = chunk * CHUNK + tw + n0;   // window index for M = n0
            if (!hi && w < NWIN) out[(size_t)row * TLEN + WSZ + w] = val;
            // head: out[:,0..WSZ-1] = f(window 0); lane 0 of wave 0, first tile
            if (p == 0 && tp == 0 && chunk == 0 && tid == 0) {
#pragma unroll
                for (int jj = 0; jj < WSZ; ++jj) out[(size_t)row * TLEN + jj] = val;
            }
        }
    }
}

extern "C" void kernel_launch(void* const* d_in, const int* in_sizes, int n_in,
                              void* d_out, int out_size, void* d_ws, size_t ws_size,
                              hipStream_t stream) {
    (void)in_sizes; (void)n_in; (void)d_ws; (void)ws_size; (void)out_size;
    const float* ret = (const float*)d_in[0];
    const float* W1  = (const float*)d_in[1];
    const float* b1  = (const float*)d_in[2];
    const float* W2  = (const float*)d_in[3];
    const float* b2  = (const float*)d_in[4];
    const float* W3  = (const float*)d_in[5];
    const float* b3  = (const float*)d_in[6];
    float* out = (float*)d_out;

    dim3 grid(BATCH * NCHUNK);   // 8192 blocks: (row, chunk)
    dim3 block(256);             // 8 wave32s
    hurst_mlp_wmma<<<grid, block, 0, stream>>>(ret, W1, b1, W2, b2, W3, b3, out);
}
